// IGAE_encoder_22454089023912
// MI455X (gfx1250) — matrix operand order repaired and checked by reading, not compiled, not run on hardware
//
#include <hip/hip_runtime.h>
#include <math.h>

// ---------------------------------------------------------------------------
// IGAE encoder for MI455X (gfx1250, wave32, WMMA)
//   z1 = spmm(A, tanh(x @ W1));  z2 = spmm(A, tanh(z1 @ W2))
//   z  = spmm(A, z2 @ W3);       adj = sigmoid(z @ z^T)
// GEMMs: bf16 inputs, f32 accumulate via v_wmma_f32_16x16x32_bf16.
// spmm:  f32 scatter, one wave per edge, global_atomic_add_f32.
// Transcendentals: v_exp_f32 + v_rcp_f32 fast paths (no IEEE div sequence).
// ---------------------------------------------------------------------------

typedef __attribute__((ext_vector_type(16))) __bf16 v16bf;
typedef __attribute__((ext_vector_type(8)))  float  v8f;

#define N_NODES 12288
#define N_EDGES 393216

__device__ __forceinline__ unsigned short f2bf(float f) {
  union { float f; unsigned int u; } v; v.f = f;
  unsigned int u = v.u;
  u += 0x7FFFu + ((u >> 16) & 1u);   // round-to-nearest-even
  return (unsigned short)(u >> 16);
}

// sigmoid(x) = rcp(1 + exp(-x));  v_mul + v_exp + v_add + v_rcp
__device__ __forceinline__ float fast_sigmoid(float x) {
  return __builtin_amdgcn_rcpf(1.0f + __expf(-x));
}

// tanh(x) = 1 - 2*rcp(exp(2x) + 1); saturates via inf/0 through v_rcp
__device__ __forceinline__ float fast_tanh(float x) {
  return 1.0f - 2.0f * __builtin_amdgcn_rcpf(__expf(2.0f * x) + 1.0f);
}

__global__ void cvt_f32_bf16(const float* __restrict__ in,
                             unsigned short* __restrict__ out, int n) {
  int i = blockIdx.x * blockDim.x + threadIdx.x;
  if (i < n) out[i] = f2bf(in[i]);
}

// Wt[m*K + k] = bf16(W[k*M + m])   (W is [K,M] row-major f32)
__global__ void cvt_transpose_bf16(const float* __restrict__ W,
                                   unsigned short* __restrict__ Wt,
                                   int K, int M) {
  int i = blockIdx.x * blockDim.x + threadIdx.x;
  if (i >= K * M) return;
  int m = i / K, k = i - m * K;
  Wt[i] = f2bf(W[(size_t)k * M + m]);
}

union bf_frag { v16bf v; unsigned short s[16]; };

// Build A fragment (16x32 bf16, ISA 7.12.2 layout) from a row-major row ptr.
__device__ __forceinline__ void load_a_frag(bf_frag& f,
                                            const unsigned short* arow,
                                            int k0, int g) {
  const unsigned short* a0 = arow + k0 + 8 * g;        // K = 8g .. 8g+7
  const unsigned short* a1 = arow + k0 + 16 + 8 * g;   // K = 16+8g .. 23+8g
#pragma unroll
  for (int h = 0; h < 8; ++h) {
    f.s[h]     = a0[h];
    f.s[h + 8] = a1[h];
  }
}

// Build B fragment (32x16 bf16) from B^T row-major row ptr (contiguous K).
__device__ __forceinline__ void load_b_frag(bf_frag& f,
                                            const unsigned short* brow,
                                            int k0, int g) {
  const unsigned short* b0 = brow + k0 + 8 * g;        // K = 8g .. 8g+7
  const unsigned short* b1 = brow + k0 + 16 + 8 * g;   // K = 16+8g .. 23+8g
#pragma unroll
  for (int h = 0; h < 8; ++h) {
    f.s[h]     = b0[h];
    f.s[h + 8] = b1[h];
  }
}

// ---------------------------------------------------------------------------
// C[Nr,M] = act(A[Nr,K] @ B[K,M]) with B supplied transposed as Bt[M,K].
// One wave per 16x16 output tile; K stepped by 32 through WMMA bf16.
// ---------------------------------------------------------------------------
template <int ACT_TANH>
__global__ __launch_bounds__(256) void gemm_bf16_wmma(
    const unsigned short* __restrict__ A,   // [Nr,K] bf16 row-major
    const unsigned short* __restrict__ Bt,  // [M,K]  bf16 row-major (B^T)
    float* __restrict__ C,                  // [Nr,M] f32
    int Nr, int K, int M) {
  int wave = (blockIdx.x * blockDim.x + threadIdx.x) >> 5;
  int lane = threadIdx.x & 31;
  int tilesM = M >> 4;
  int tilesN = Nr >> 4;
  if (wave >= tilesM * tilesN) return;      // wave-uniform: EXEC stays all-1s
  int tm = wave / tilesM;
  int tn = wave - tm * tilesM;
  int g = lane >> 4;                        // lane group (0 or 1)
  int l = lane & 15;

  const unsigned short* arow = A  + (size_t)(tm * 16 + l) * K;
  const unsigned short* brow = Bt + (size_t)(tn * 16 + l) * K;

  v8f acc = {};
  for (int k0 = 0; k0 < K; k0 += 32) {
    bf_frag fa, fb;
    load_a_frag(fa, arow, k0, g);
    load_b_frag(fb, brow, k0, g);
    acc = __builtin_amdgcn_wmma_f32_16x16x32_bf16(
        false, fa.v, false, fb.v, (short)0, acc, false, false);
  }

#pragma unroll
  for (int v = 0; v < 8; ++v) {
    int r = tm * 16 + v + 8 * g;
    int c = tn * 16 + l;
    float val = acc[v];
    if (ACT_TANH) val = fast_tanh(val);
    C[(size_t)r * M + c] = val;
  }
}

// ---------------------------------------------------------------------------
// adj[r,c] = sigmoid( sum_k z[r,k] * z[c,k] ), z in bf16 [N,64].
// One wave per 16x64 output tile: A fragments (K=64 -> 2) loaded once and
// reused across 4 column sub-tiles; B = z^T so B fragments read rows of z.
// ---------------------------------------------------------------------------
__global__ __launch_bounds__(256) void adj_sigmoid_wmma(
    const unsigned short* __restrict__ Zb,  // [N,64] bf16
    float* __restrict__ out) {              // [N,N] f32
  int wave = (blockIdx.x * blockDim.x + threadIdx.x) >> 5;
  int lane = threadIdx.x & 31;
  const int rowTiles = N_NODES / 16;        // 768
  const int colGrps  = N_NODES / 64;        // 192
  if (wave >= rowTiles * colGrps) return;
  int tm = wave / colGrps;
  int tc = wave - tm * colGrps;
  int g = lane >> 4;
  int l = lane & 15;

  const unsigned short* arow = Zb + (size_t)(tm * 16 + l) * 64;
  bf_frag fa0, fa1;
  load_a_frag(fa0, arow, 0, g);
  load_a_frag(fa1, arow, 32, g);

#pragma unroll
  for (int j = 0; j < 4; ++j) {
    int colBase = tc * 64 + j * 16;
    const unsigned short* brow = Zb + (size_t)(colBase + l) * 64;
    bf_frag fb0, fb1;
    load_b_frag(fb0, brow, 0, g);
    load_b_frag(fb1, brow, 32, g);
    v8f acc = {};
    acc = __builtin_amdgcn_wmma_f32_16x16x32_bf16(
        false, fa0.v, false, fb0.v, (short)0, acc, false, false);
    acc = __builtin_amdgcn_wmma_f32_16x16x32_bf16(
        false, fa1.v, false, fb1.v, (short)0, acc, false, false);
#pragma unroll
    for (int v = 0; v < 8; ++v) {
      int r = tm * 16 + v + 8 * g;
      int c = colBase + l;
      out[(size_t)r * N_NODES + c] = fast_sigmoid(acc[v]);
    }
  }
}

// ---------------------------------------------------------------------------
// out[row[e], f] += w[e] * H[col[e], f]   (f32 L2-side atomics)
// One wave per edge; each lane covers F/32 consecutive features (vector load).
// ---------------------------------------------------------------------------
template <int F>
__global__ __launch_bounds__(256) void spmm_scatter(
    const int* __restrict__ er, const int* __restrict__ ec,
    const float* __restrict__ ew, const float* __restrict__ H,
    float* __restrict__ out) {
  constexpr int C = F / 32;                 // floats per lane (2/4/8)
  int wave = (blockIdx.x * blockDim.x + threadIdx.x) >> 5;
  int lane = threadIdx.x & 31;
  if (wave >= N_EDGES) return;
  int e = wave;
  int c = ec[e];
  int r = er[e];
  float w = ew[e];
  const float* src = H  + (size_t)c * F + lane * C;
  float*       dst = out + (size_t)r * F + lane * C;
  float v[C];
#pragma unroll
  for (int i = 0; i < C; ++i) v[i] = src[i];   // merges into b128/b64 loads
#pragma unroll
  for (int i = 0; i < C; ++i) atomicAdd(dst + i, w * v[i]);
}

// ---------------------------------------------------------------------------
extern "C" void kernel_launch(void* const* d_in, const int* in_sizes, int n_in,
                              void* d_out, int out_size, void* d_ws, size_t ws_size,
                              hipStream_t stream) {
  const float* x  = (const float*)d_in[0];
  const int*   er = (const int*)d_in[1];
  const int*   ec = (const int*)d_in[2];
  const float* ew = (const float*)d_in[3];
  const float* W1 = (const float*)d_in[4];
  const float* W2 = (const float*)d_in[5];
  const float* W3 = (const float*)d_in[6];

  float* out_z   = (float*)d_out;                       // [N,64]
  float* out_adj = out_z + (size_t)N_NODES * 64;        // [N,N]

  // workspace carve-out (256B aligned)
  char* p = (char*)d_ws;
  auto carve = [&](size_t bytes) {
    char* r = p;
    p += (bytes + 255) & ~(size_t)255;
    return (void*)r;
  };
  unsigned short* xb  = (unsigned short*)carve((size_t)N_NODES * 512 * 2);
  unsigned short* W1t = (unsigned short*)carve((size_t)512 * 256 * 2);
  unsigned short* W2t = (unsigned short*)carve((size_t)256 * 128 * 2);
  unsigned short* W3t = (unsigned short*)carve((size_t)128 * 64 * 2);
  float*          H1  = (float*)carve((size_t)N_NODES * 256 * 4);
  float*          z1  = (float*)carve((size_t)N_NODES * 256 * 4);
  unsigned short* z1b = (unsigned short*)carve((size_t)N_NODES * 256 * 2);
  float*          H2  = (float*)carve((size_t)N_NODES * 128 * 4);
  float*          z2  = (float*)carve((size_t)N_NODES * 128 * 4);
  unsigned short* z2b = (unsigned short*)carve((size_t)N_NODES * 128 * 2);
  float*          H3  = (float*)carve((size_t)N_NODES * 64 * 4);
  unsigned short* zb  = (unsigned short*)carve((size_t)N_NODES * 64 * 2);

  const dim3 blk(256);
  auto blocks = [](long long threads) { return (unsigned)((threads + 255) / 256); };

  // --- input conversions -------------------------------------------------
  cvt_f32_bf16<<<blocks((long long)N_NODES * 512), blk, 0, stream>>>(
      x, xb, N_NODES * 512);
  cvt_transpose_bf16<<<blocks(512 * 256), blk, 0, stream>>>(W1, W1t, 512, 256);
  cvt_transpose_bf16<<<blocks(256 * 128), blk, 0, stream>>>(W2, W2t, 256, 128);
  cvt_transpose_bf16<<<blocks(128 * 64),  blk, 0, stream>>>(W3, W3t, 128, 64);

  // --- layer 1: H1 = tanh(x @ W1); z1 = spmm(H1) -------------------------
  {
    long long waves = (long long)(N_NODES / 16) * (256 / 16);
    gemm_bf16_wmma<1><<<blocks(waves * 32), blk, 0, stream>>>(
        xb, W1t, H1, N_NODES, 512, 256);
  }
  hipMemsetAsync(z1, 0, (size_t)N_NODES * 256 * 4, stream);
  spmm_scatter<256><<<blocks((long long)N_EDGES * 32), blk, 0, stream>>>(
      er, ec, ew, H1, z1);
  cvt_f32_bf16<<<blocks((long long)N_NODES * 256), blk, 0, stream>>>(
      z1, z1b, N_NODES * 256);

  // --- layer 2: H2 = tanh(z1 @ W2); z2 = spmm(H2) ------------------------
  {
    long long waves = (long long)(N_NODES / 16) * (128 / 16);
    gemm_bf16_wmma<1><<<blocks(waves * 32), blk, 0, stream>>>(
        z1b, W2t, H2, N_NODES, 256, 128);
  }
  hipMemsetAsync(z2, 0, (size_t)N_NODES * 128 * 4, stream);
  spmm_scatter<128><<<blocks((long long)N_EDGES * 32), blk, 0, stream>>>(
      er, ec, ew, H2, z2);
  cvt_f32_bf16<<<blocks((long long)N_NODES * 128), blk, 0, stream>>>(
      z2, z2b, N_NODES * 128);

  // --- layer 3 (no activation): H3 = z2 @ W3; z = spmm(H3) ---------------
  {
    long long waves = (long long)(N_NODES / 16) * (64 / 16);
    gemm_bf16_wmma<0><<<blocks(waves * 32), blk, 0, stream>>>(
        z2b, W3t, H3, N_NODES, 128, 64);
  }
  hipMemsetAsync(out_z, 0, (size_t)N_NODES * 64 * 4, stream);
  spmm_scatter<64><<<blocks((long long)N_EDGES * 32), blk, 0, stream>>>(
      er, ec, ew, H3, out_z);

  // --- adjacency reconstruction: sigmoid(z @ z^T) ------------------------
  cvt_f32_bf16<<<blocks((long long)N_NODES * 64), blk, 0, stream>>>(
      out_z, zb, N_NODES * 64);
  {
    long long waves = (long long)(N_NODES / 16) * (N_NODES / 64);
    adj_sigmoid_wmma<<<blocks(waves * 32), blk, 0, stream>>>(zb, out_adj);
  }
}
